// DeepseekV3MoE_71803263255218
// MI455X (gfx1250) — compile-verified, matrix operand order checked by reading, not run on hardware
//
#include <hip/hip_runtime.h>
#include <hip/hip_bf16.h>
#include <math.h>

// ---------------------------------------------------------------------------
// DeepSeek-V3 MoE for MI455X (gfx1250), wave32 + WMMA bf16.
// T=4096 tokens, H=1024, E=64 experts, F=512, K=8, G=8 groups, TG=4, scale 2.5
// ---------------------------------------------------------------------------

#define T_TOK   4096
#define HID     1024
#define NEXP    64
#define FINT    512
#define TOPK    8
#define NGRP    8
#define TGRP    4
#define RSCALE  2.5f

// GEMM tiling
#define TM   128          // tokens per tile
#define TN   128          // F-chunk / H-chunk width
#define KC   64           // K staging chunk
#define XS   72           // sX row stride (elements) -> 144B, 16B aligned
#define WS   72           // sW row stride
#define HS   520          // sHm row stride -> 1040B, 16B aligned

typedef __attribute__((ext_vector_type(16))) __bf16 v16bf;
typedef __attribute__((ext_vector_type(8)))  float  v8f;

union ABfrag {
    v16bf bf;
    unsigned int u[8];
    uint4 q[2];
};

__device__ __forceinline__ unsigned short f2bf(float f) {
    unsigned int u = __float_as_uint(f);
    unsigned int r = u + 0x7FFFu + ((u >> 16) & 1u);   // round-to-nearest-even
    return (unsigned short)(r >> 16);
}

__device__ __forceinline__ v8f v8f_zero() {
    v8f z;
#pragma unroll
    for (int i = 0; i < 8; ++i) z[i] = 0.f;
    return z;
}

__device__ __forceinline__ void ld_frag(ABfrag& f, const unsigned short* p) {
    f.q[0] = *(const uint4*)p;
    f.q[1] = *(const uint4*)(p + 16);
}

// ---------------------------------------------------------------------------
// Kernel 0: zero output accumulator + per-expert counters
// ---------------------------------------------------------------------------
__global__ void zero_kernel(float* __restrict__ y, int* __restrict__ cnt) {
    int idx = blockIdx.x * blockDim.x + threadIdx.x;   // 4096 blocks * 256 thr
    float4 z = make_float4(0.f, 0.f, 0.f, 0.f);
    ((float4*)y)[idx] = z;                             // covers T_TOK*HID floats
    if (blockIdx.x == 0 && threadIdx.x < NEXP) cnt[threadIdx.x] = 0;
}

// ---------------------------------------------------------------------------
// Kernel 1: gate (sigmoid scores, group top-2 sum, top-4 groups, top-8 experts)
// one block (64 threads) per token; compacts token into per-expert lists.
// ---------------------------------------------------------------------------
__global__ void gate_kernel(const float* __restrict__ x,
                            const float* __restrict__ gw,
                            const float* __restrict__ gb,
                            int*   __restrict__ cnt,
                            int*   __restrict__ tok_list,
                            float* __restrict__ w_list) {
    const int t   = blockIdx.x;
    const int tid = threadIdx.x;     // 64 threads, expert id == tid
    __shared__ float sx[HID];
    __shared__ float s_sc[NEXP];
    __shared__ float s_cor[NEXP];

    for (int i = tid; i < HID; i += 64) sx[i] = x[(size_t)t * HID + i];
    __syncthreads();

    const float* wr = gw + (size_t)tid * HID;
    float acc = 0.f;
#pragma unroll 4
    for (int h = 0; h < HID; ++h) acc += sx[h] * wr[h];
    float sc = 1.f / (1.f + __expf(-acc));
    s_sc[tid]  = sc;
    s_cor[tid] = sc + gb[tid];
    __syncthreads();

    if (tid == 0) {
        // group scores = sum of top-2 corrected scores within group
        float gs[NGRP];
#pragma unroll
        for (int g = 0; g < NGRP; ++g) {
            float m1 = -1e30f, m2 = -1e30f;
#pragma unroll
            for (int j = 0; j < NEXP / NGRP; ++j) {
                float v = s_cor[g * (NEXP / NGRP) + j];
                if (v > m1) { m2 = m1; m1 = v; }
                else if (v > m2) { m2 = v; }
            }
            gs[g] = m1 + m2;
        }
        // top-4 groups
        unsigned gmask = 0;
#pragma unroll
        for (int j = 0; j < TGRP; ++j) {
            int best = 0; float bv = -1e30f;
            for (int g = 0; g < NGRP; ++g)
                if (!((gmask >> g) & 1u) && gs[g] > bv) { bv = gs[g]; best = g; }
            gmask |= 1u << best;
        }
        // top-8 experts among allowed groups by corrected score
        int   tix[TOPK];
        float tw[TOPK];
        float wsum = 0.f;
        unsigned long long used = 0ull;
#pragma unroll
        for (int k = 0; k < TOPK; ++k) {
            int best = 0; float bv = -1e30f;
            for (int e = 0; e < NEXP; ++e) {
                if (((gmask >> (e >> 3)) & 1u) && !((used >> e) & 1ull)) {
                    float v = s_cor[e];
                    if (v > bv) { bv = v; best = e; }
                }
            }
            used |= 1ull << best;
            tix[k] = best;
            tw[k]  = s_sc[best];          // weights from ORIGINAL scores
            wsum  += tw[k];
        }
        float inv = RSCALE / (wsum + 1e-20f);
#pragma unroll
        for (int k = 0; k < TOPK; ++k) {
            int e   = tix[k];
            int pos = atomicAdd(&cnt[e], 1);
            tok_list[e * T_TOK + pos] = t;
            w_list [e * T_TOK + pos] = tw[k] * inv;
        }
    }
}

// ---------------------------------------------------------------------------
// Kernel 2/3: expert MLP  y += cw * ( silu(X Wg) * (X Wu) ) Wd
// 256 threads = 8 waves; each wave owns one 16-row M strip.
// LDS (185KB) limits us to 1 block/WGP = 2 waves/SIMD, so declare
// waves_per_eu(2,8) to unlock the RF/2 VGPR budget for fragment double-buffers.
// ROUTED: gathered token tile, atomic accumulate. !ROUTED: shared expert.
// ---------------------------------------------------------------------------
template <bool ROUTED>
__global__ __launch_bounds__(256)
__attribute__((amdgpu_waves_per_eu(2, 8)))
void moe_mlp_kernel(const float* __restrict__ x,
                    const float* __restrict__ Wg_all,   // [H,F] per expert
                    const float* __restrict__ Wu_all,   // [H,F]
                    const float* __restrict__ Wd_all,   // [F,H]
                    float* __restrict__ y,
                    const int*   __restrict__ cnt,
                    const int*   __restrict__ tok_list,
                    const float* __restrict__ w_list) {
    extern __shared__ unsigned char smem[];
    int*            stok = (int*)smem;                       // [TM]
    float*          swt  = (float*)(smem + 512);             // [TM]
    unsigned short* sX   = (unsigned short*)(smem + 1024);   // [TM][XS]
    unsigned short* sWg  = sX  + TM * XS;                    // [TN][WS] (n-major)
    unsigned short* sWu  = sWg + TN * WS;                    // [TN][WS]
    unsigned short* sHm  = sWu + TN * WS;                    // [TM][HS] bf16 hmid

    const int tid  = threadIdx.x;
    const int lane = tid & 31;
    const int wv   = tid >> 5;                 // 0..7 -> M tile index

    const float *Wg, *Wu, *Wd;
    int tile, nrows, e = 0;
    if constexpr (ROUTED) {
        e    = blockIdx.x;
        tile = blockIdx.y;
        int n = cnt[e];
        if (tile * TM >= n) return;            // uniform per block
        nrows = n - tile * TM; if (nrows > TM) nrows = TM;
        Wg = Wg_all + (size_t)e * HID * FINT;
        Wu = Wu_all + (size_t)e * HID * FINT;
        Wd = Wd_all + (size_t)e * FINT * HID;
    } else {
        tile  = blockIdx.x;
        nrows = TM;
        Wg = Wg_all; Wu = Wu_all; Wd = Wd_all;
    }

    if (tid < TM) {
        if constexpr (ROUTED) {
            if (tid < nrows) {
                stok[tid] = tok_list[e * T_TOK + tile * TM + tid];
                swt [tid] = w_list [e * T_TOK + tile * TM + tid];
            } else { stok[tid] = -1; swt[tid] = 0.f; }
        } else {
            stok[tid] = tile * TM + tid;
            swt [tid] = 1.f;
        }
    }

    const int off0 = (lane < 16) ? 0 : 8;      // bf16 K-pair half per lane group
    const int l15  = lane & 15;

    // ----------------- phase 1: hmid = silu(X Wg) * (X Wu)  -> sHm -----------
    for (int nf = 0; nf < FINT / TN; ++nf) {
        v8f accG[8], accU[8];
#pragma unroll
        for (int i = 0; i < 8; ++i) { accG[i] = v8f_zero(); accU[i] = v8f_zero(); }

        for (int kc = 0; kc < HID / KC; ++kc) {
            __syncthreads();
            // stage X (gathered rows, fp32 -> bf16)
            {
                int m  = tid >> 1;
                int hh = (tid & 1) * 32;
                int tgt = stok[m];
                unsigned short* dst = sX + m * XS + hh;
                if (tgt >= 0) {
                    const float* src = x + (size_t)tgt * HID + kc * KC + hh;
#pragma unroll
                    for (int i = 0; i < 32; i += 4) {
                        float4 v = *(const float4*)(src + i);
                        dst[i + 0] = f2bf(v.x); dst[i + 1] = f2bf(v.y);
                        dst[i + 2] = f2bf(v.z); dst[i + 3] = f2bf(v.w);
                    }
                } else {
#pragma unroll
                    for (int i = 0; i < 32; ++i) dst[i] = 0;
                }
            }
            // stage Wg/Wu transposed: sW[n][k] = W[(kc*KC+k)*F + nf*TN + n]
            for (int idx = tid; idx < KC * TN; idx += 256) {
                int k = idx >> 7, n = idx & (TN - 1);
                size_t go = (size_t)(kc * KC + k) * FINT + nf * TN + n;
                sWg[n * WS + k] = f2bf(Wg[go]);
                sWu[n * WS + k] = f2bf(Wu[go]);
            }
            // prefetch next K chunk of the weight stream
            if (kc + 1 < HID / KC) {
                size_t pn = (size_t)((kc + 1) * KC + (tid >> 2)) * FINT + nf * TN + (tid & 3) * 32;
                __builtin_prefetch(Wg + pn, 0, 1);
                __builtin_prefetch(Wu + pn, 0, 1);
            }
            __syncthreads();

            // ---- software-pipelined WMMA: preload A (both K-steps) + B dbl-buf
            ABfrag a[2];
            {
                const unsigned short* p = sX + (wv * 16 + l15) * XS + off0;
                ld_frag(a[0], p);
                ld_frag(a[1], p + 32);
            }
            ABfrag bg[2], bu[2];
            ld_frag(bg[0], sWg + l15 * WS + off0);
            ld_frag(bu[0], sWu + l15 * WS + off0);
#pragma unroll
            for (int j = 0; j < 16; ++j) {
                const int ks  = j >> 3;
                const int n   = j & 7;
                const int cur = j & 1;
                if (j < 15) {
                    const int jn  = j + 1;
                    const int ksn = jn >> 3;
                    const int nn  = jn & 7;
                    ld_frag(bg[cur ^ 1], sWg + (nn * 16 + l15) * WS + ksn * 32 + off0);
                    ld_frag(bu[cur ^ 1], sWu + (nn * 16 + l15) * WS + ksn * 32 + off0);
                }
                accG[n] = __builtin_amdgcn_wmma_f32_16x16x32_bf16(
                    false, a[ks].bf, false, bg[cur].bf, (short)0, accG[n], false, false);
                accU[n] = __builtin_amdgcn_wmma_f32_16x16x32_bf16(
                    false, a[ks].bf, false, bu[cur].bf, (short)0, accU[n], false, false);
            }
        }
        // SwiGLU on f32 accumulators -> bf16 in sHm
        {
            int mrow0 = wv * 16 + ((lane >> 4) << 3);
            int ncol0 = nf * TN + l15;
#pragma unroll
            for (int n = 0; n < 8; ++n) {
#pragma unroll
                for (int i = 0; i < 8; ++i) {
                    float g = accG[n][i], u = accU[n][i];
                    float h = (g / (1.f + __expf(-g))) * u;   // silu(g)*u
                    sHm[(mrow0 + i) * HS + ncol0 + n * 16] = f2bf(h);
                }
            }
        }
    }
    __syncthreads();

    // ----------------- phase 2: y += cw * (hmid Wd) ---------------------------
    for (int nh = 0; nh < HID / TN; ++nh) {
        v8f accD[8];
#pragma unroll
        for (int i = 0; i < 8; ++i) accD[i] = v8f_zero();

        for (int kc = 0; kc < FINT / KC; ++kc) {
            __syncthreads();
            // stage Wd chunk (reuse sWg): sWg[n][k] = Wd[(kc*KC+k)*H + nh*TN + n]
            for (int idx = tid; idx < KC * TN; idx += 256) {
                int k = idx >> 7, n = idx & (TN - 1);
                size_t go = (size_t)(kc * KC + k) * HID + nh * TN + n;
                sWg[n * WS + k] = f2bf(Wd[go]);
            }
            if (kc + 1 < FINT / KC) {
                size_t pn = (size_t)((kc + 1) * KC + (tid >> 2)) * HID + nh * TN + (tid & 3) * 32;
                __builtin_prefetch(Wd + pn, 0, 1);
            }
            __syncthreads();

            // ---- software-pipelined WMMA over hmid (A from sHm, B dbl-buf)
            ABfrag a[2];
            {
                const unsigned short* p = sHm + (wv * 16 + l15) * HS + kc * KC + off0;
                ld_frag(a[0], p);
                ld_frag(a[1], p + 32);
            }
            ABfrag b[2];
            ld_frag(b[0], sWg + l15 * WS + off0);
#pragma unroll
            for (int j = 0; j < 16; ++j) {
                const int ks  = j >> 3;
                const int n   = j & 7;
                const int cur = j & 1;
                if (j < 15) {
                    const int jn  = j + 1;
                    const int ksn = jn >> 3;
                    const int nn  = jn & 7;
                    ld_frag(b[cur ^ 1], sWg + (nn * 16 + l15) * WS + ksn * 32 + off0);
                }
                accD[n] = __builtin_amdgcn_wmma_f32_16x16x32_bf16(
                    false, a[ks].bf, false, b[cur].bf, (short)0, accD[n], false, false);
            }
        }
        // scale by routing weight and accumulate into y
        {
            int mrow0 = wv * 16 + ((lane >> 4) << 3);
            int ncb   = nh * TN + l15;
#pragma unroll
            for (int n = 0; n < 8; ++n) {
#pragma unroll
                for (int i = 0; i < 8; ++i) {
                    int row = mrow0 + i;
                    int tgt = stok[row];
                    if (tgt >= 0) {
                        float v = accD[n][i] * swt[row];
                        float* dst = y + (size_t)tgt * HID + ncb + n * 16;
                        if constexpr (ROUTED) {
                            unsafeAtomicAdd(dst, v);      // global_atomic_add_f32
                        } else {
                            *dst += v;                    // runs after routed kernels
                        }
                    }
                }
            }
        }
    }
}

// ---------------------------------------------------------------------------
// Launch
// ---------------------------------------------------------------------------
extern "C" void kernel_launch(void* const* d_in, const int* in_sizes, int n_in,
                              void* d_out, int out_size, void* d_ws, size_t ws_size,
                              hipStream_t stream) {
    const float* x   = (const float*)d_in[0];   // [B,S,H]
    const float* gw  = (const float*)d_in[1];   // [E,H]
    const float* gb  = (const float*)d_in[2];   // [E]
    const float* Weg = (const float*)d_in[3];   // [E,H,F]
    const float* Weu = (const float*)d_in[4];   // [E,H,F]
    const float* Wed = (const float*)d_in[5];   // [E,F,H]
    const float* Wsg = (const float*)d_in[6];   // [H,F_SH]
    const float* Wsu = (const float*)d_in[7];   // [H,F_SH]
    const float* Wsd = (const float*)d_in[8];   // [F_SH,H]
    float* y = (float*)d_out;                   // [B,S,H]

    int*   cnt      = (int*)d_ws;
    int*   tok_list = (int*)((char*)d_ws + 256);
    float* w_list   = (float*)((char*)d_ws + 256 + (size_t)NEXP * T_TOK * sizeof(int));

    const size_t smem_bytes = 1024                                   // stok + swt
                            + (size_t)TM * XS * sizeof(unsigned short)
                            + 2 * (size_t)TN * WS * sizeof(unsigned short)
                            + (size_t)TM * HS * sizeof(unsigned short);   // ~185KB (WGP has 320KB)

    // zero y and counters
    zero_kernel<<<(T_TOK * HID / 4) / 256, 256, 0, stream>>>(y, cnt);
    // gate + routing compaction
    gate_kernel<<<T_TOK, 64, 0, stream>>>(x, gw, gb, cnt, tok_list, w_list);
    // routed experts: grid = (expert, token-tile); empty tiles exit immediately
    moe_mlp_kernel<true><<<dim3(NEXP, T_TOK / TM), 256, smem_bytes, stream>>>(
        x, Weg, Weu, Wed, y, cnt, tok_list, w_list);
    // shared expert over all tokens (stream-ordered after routed accumulation)
    moe_mlp_kernel<false><<<dim3(T_TOK / TM, 1), 256, smem_bytes, stream>>>(
        x, Wsg, Wsu, Wsd, y, nullptr, nullptr, nullptr);
}
